// MultiheadSelfAttention_81432579932848
// MI455X (gfx1250) — matrix operand order repaired and checked by reading, not compiled, not run on hardware
//
#include <hip/hip_runtime.h>

typedef __attribute__((ext_vector_type(16))) __bf16 v16bf;
typedef __attribute__((ext_vector_type(8)))  __bf16 v8bf;
typedef __attribute__((ext_vector_type(8)))  float  v8f;
typedef __attribute__((ext_vector_type(4)))  int    v4i;

#define D_MODEL 1024
#define NHEADS  16
#define DK      64
#define SEQ     2048
#define BATCH   4
#define MTOK    (BATCH * SEQ)   // 8192

static __device__ __forceinline__ __bf16 f2bf(float f) {
    unsigned int u = __builtin_bit_cast(unsigned int, f);
    u += 0x7FFFu + ((u >> 16) & 1u);           // round-to-nearest-even
    unsigned short h = (unsigned short)(u >> 16);
    return __builtin_bit_cast(__bf16, h);
}
static __device__ __forceinline__ float bf2f(__bf16 h) {
    unsigned short u = __builtin_bit_cast(unsigned short, h);
    unsigned int w = ((unsigned int)u) << 16;
    return __builtin_bit_cast(float, w);
}
static __device__ __forceinline__ v8bf ld8(const __bf16* p) {
    return *(const v8bf*)p;                    // 16B, caller guarantees alignment
}
static __device__ __forceinline__ v16bf mk16(v8bf lo, v8bf hi) {
    return __builtin_shufflevector(lo, hi, 0,1,2,3,4,5,6,7,8,9,10,11,12,13,14,15);
}
static __device__ __forceinline__ v16bf ld16(const __bf16* p) {
    return mk16(ld8(p), ld8(p + 8));
}
static __device__ __forceinline__ v8f wmma_bf16(v16bf a, v16bf b, v8f c) {
    return __builtin_amdgcn_wmma_f32_16x16x32_bf16(false, a, false, b, (short)0, c, false, false);
}
// async copy global -> LDS, 16B per lane, tracked by ASYNCcnt
static __device__ __forceinline__ void async_cp16(void* lds, const void* g) {
    __builtin_amdgcn_global_load_async_to_lds_b128((v4i*)g, (v4i*)lds, 0, 0);
}

// ---------------------------------------------------------------- convert
__global__ void cvt_f32_bf16(const float* __restrict__ src, __bf16* __restrict__ dst, int n) {
    for (int i = blockIdx.x * blockDim.x + threadIdx.x; i < n; i += gridDim.x * blockDim.x)
        dst[i] = f2bf(src[i]);
}

// ---------------------------------------------------------------- GEMM
// C[m,n] = sum_k A[m,k] * W[n,k].  A: [M x K] bf16 row-major, W: [N x K] bf16 row-major.
// mode 0: store fp32 to outF[m*N+n]
// mode 1: (QKV) n -> (g,h,d); scatter bf16 into qb/kb/vb laid out [B,H,S,DK]
__global__ __launch_bounds__(256) void gemm_bf16(
    const __bf16* __restrict__ A, const __bf16* __restrict__ W,
    float* __restrict__ outF, __bf16* __restrict__ qb, __bf16* __restrict__ kb,
    __bf16* __restrict__ vb, int N, int K, int mode)
{
    __shared__ __bf16 At[2][128][40];   // double-buffered 128 x 32 (+pad)
    __shared__ __bf16 Bt[2][64][40];    // double-buffered 64  x 32 (+pad)

    const int tid  = threadIdx.x;
    const int wave = tid >> 5, lane = tid & 31;
    const int ln15 = lane & 15, hlf = lane >> 4;
    const int wm = wave & 3, wn = wave >> 2;
    const int M0 = blockIdx.y * 128, N0 = blockIdx.x * 64;
    const int kb8 = hlf ? 8 : 0, kb16 = hlf ? 16 : 0;

    v8f acc[2][2] = {};

    auto stage = [&](int k0, int buf) {
        #pragma unroll
        for (int it = 0; it < 2; ++it) {
            int cc = tid + it * 256;
            int row = cc >> 2, kc = (cc & 3) * 8;
            async_cp16(&At[buf][row][kc], A + (size_t)(M0 + row) * K + k0 + kc);
        }
        int n = tid >> 2, kc = (tid & 3) * 8;
        async_cp16(&Bt[buf][n][kc], W + (size_t)(N0 + n) * K + k0 + kc);
    };

    const int nk = K >> 5;
    stage(0, 0);
    for (int i = 0; i < nk; ++i) {
        const int buf = i & 1;
        if (i + 1 < nk) {
            stage((i + 1) << 5, buf ^ 1);
            asm volatile("s_wait_asynccnt 0x3" ::: "memory");  // tile i landed (in-order)
        } else {
            asm volatile("s_wait_asynccnt 0x0" ::: "memory");
        }
        __syncthreads();

        v16bf a[2], b[2];
        #pragma unroll
        for (int mt = 0; mt < 2; ++mt) {
            int row = wm * 32 + mt * 16 + ln15;
            a[mt] = mk16(ld8(&At[buf][row][kb8]), ld8(&At[buf][row][16 + kb8]));
        }
        #pragma unroll
        for (int nt = 0; nt < 2; ++nt) {
            int col = wn * 32 + nt * 16 + ln15;
            b[nt] = mk16(ld8(&Bt[buf][col][kb8]), ld8(&Bt[buf][col][16 + kb8]));
        }
        #pragma unroll
        for (int mt = 0; mt < 2; ++mt)
            #pragma unroll
            for (int nt = 0; nt < 2; ++nt)
                acc[mt][nt] = wmma_bf16(a[mt], b[nt], acc[mt][nt]);

        __syncthreads();   // all waves done with buf before it is overwritten
    }

    #pragma unroll
    for (int mt = 0; mt < 2; ++mt)
        #pragma unroll
        for (int nt = 0; nt < 2; ++nt)
            #pragma unroll
            for (int r = 0; r < 8; ++r) {
                int m = M0 + wm * 32 + mt * 16 + r + hlf * 8;
                int n = N0 + wn * 32 + nt * 16 + ln15;
                float v = acc[mt][nt][r];
                if (mode == 0) {
                    outF[(size_t)m * N + n] = v;
                } else {
                    int g = n >> 10, rem = n & 1023;
                    int hh = rem >> 6, dd = rem & 63;
                    int bb = m >> 11, ss = m & 2047;
                    __bf16* dst = (g == 0) ? qb : ((g == 1) ? kb : vb);
                    dst[(((size_t)bb * NHEADS + hh) * SEQ + ss) * DK + dd] = f2bf(v);
                }
            }
}

// ---------------------------------------------------------------- RoPE (in-place on [B,H,S,DK] bf16)
__global__ void rope_kernel(__bf16* __restrict__ buf, const int* __restrict__ tp) {
    int idx = blockIdx.x * blockDim.x + threadIdx.x;   // B*H*S*32 threads
    int j  = idx & 31;
    int s  = (idx >> 5) & (SEQ - 1);
    int bh = idx >> 16;                                // SEQ*32 == 1<<16
    float pos = (float)tp[s];
    float inv_freq = __powf(10000.0f, -((float)(2 * j)) * (1.0f / 64.0f));
    float ang = pos * inv_freq;
    float sn, cn;
    __sincosf(ang, &sn, &cn);
    size_t base = ((size_t)bh * SEQ + s) * DK + 2 * j;
    float e = bf2f(buf[base]);
    float o = bf2f(buf[base + 1]);
    buf[base]     = f2bf(cn * e - sn * o);
    buf[base + 1] = f2bf(sn * e + cn * o);
}

// ---------------------------------------------------------------- flash attention
// grid: (S/128, B*H); block 256 = 8 waves; wave w owns 16 query rows.
__global__ __launch_bounds__(256) void attn_kernel(
    const __bf16* __restrict__ Q, const __bf16* __restrict__ K,
    const __bf16* __restrict__ V, __bf16* __restrict__ ctx)
{
    __shared__ __bf16 Kt[64][72];       // [key][d]
    __shared__ __bf16 Vt[64][72];       // [d][key] (transposed)
    __shared__ __bf16 Pw[8][16][72];    // per-wave P scratch [row][key]

    const int bh = blockIdx.y;
    const int b = bh >> 4, h = bh & 15;
    const size_t headOff = (size_t)bh * SEQ * DK;
    const int q0 = blockIdx.x * 128;
    const int tid = threadIdx.x, wave = tid >> 5, lane = tid & 31;
    const int ln15 = lane & 15, hlf = lane >> 4;
    const int qbase = q0 + wave * 16;
    const int kb8 = hlf ? 8 : 0, kb16 = hlf ? 16 : 0;

    // preload Q fragments (16 rows x 64 d -> two 16x32 A-frags)
    const __bf16* qrow = Q + headOff + (size_t)(qbase + ln15) * DK;
    v16bf aq0 = mk16(ld8(qrow + kb8),      ld8(qrow + 16 + kb8));
    v16bf aq1 = mk16(ld8(qrow + 32 + kb8), ld8(qrow + 48 + kb8));

    float mstate[8], lstate[8];
    v8f co[4] = {};
    #pragma unroll
    for (int r = 0; r < 8; ++r) { mstate[r] = -1e30f; lstate[r] = 0.0f; }

    const int njt = (q0 >> 6) + 2;
    for (int jt = 0; jt < njt; ++jt) {
        const int n0 = jt << 6;
        __syncthreads();
        // stage K via async-to-LDS; V through VGPRs (transposed scatter)
        v8bf vv[2];
        #pragma unroll
        for (int it = 0; it < 2; ++it) {
            int cc = tid + it * 256;
            int row = cc >> 3, dc = (cc & 7) * 8;
            async_cp16(&Kt[row][dc], K + headOff + (size_t)(n0 + row) * DK + dc);
            vv[it] = ld8(V + headOff + (size_t)(n0 + row) * DK + dc);
        }
        #pragma unroll
        for (int it = 0; it < 2; ++it) {
            int cc = tid + it * 256;
            int row = cc >> 3, dc = (cc & 7) * 8;
            #pragma unroll
            for (int j = 0; j < 8; ++j) Vt[dc + j][row] = vv[it][j];
        }
        asm volatile("s_wait_asynccnt 0x0" ::: "memory");
        __syncthreads();

        if (n0 <= qbase + 15) {
            // scores: 16 (queries) x 64 (keys) in four 16x16 C tiles
            v8f cs[4];
            #pragma unroll
            for (int nt = 0; nt < 4; ++nt) {
                int kr = nt * 16 + ln15;
                v16bf bk0 = mk16(ld8(&Kt[kr][kb8]), ld8(&Kt[kr][16 + kb8]));
                v16bf bk1 = mk16(ld8(&Kt[kr][32 + kb8]), ld8(&Kt[kr][48 + kb8]));
                v8f s = {};
                s = wmma_bf16(aq0, bk0, s);
                s = wmma_bf16(aq1, bk1, s);
                cs[nt] = s;
            }
            // causal mask + scale + row max
            float rowmax[8];
            #pragma unroll
            for (int r = 0; r < 8; ++r) rowmax[r] = -1e30f;
            #pragma unroll
            for (int nt = 0; nt < 4; ++nt) {
                int ng = n0 + nt * 16 + ln15;
                #pragma unroll
                for (int r = 0; r < 8; ++r) {
                    int mg = qbase + r + hlf * 8;
                    float v = cs[nt][r] * 0.125f;
                    v = (ng <= mg) ? v : -1e30f;
                    cs[nt][r] = v;
                    rowmax[r] = fmaxf(rowmax[r], v);
                }
            }
            #pragma unroll
            for (int r = 0; r < 8; ++r) {
                rowmax[r] = fmaxf(rowmax[r], __shfl_xor(rowmax[r], 1, 32));
                rowmax[r] = fmaxf(rowmax[r], __shfl_xor(rowmax[r], 2, 32));
                rowmax[r] = fmaxf(rowmax[r], __shfl_xor(rowmax[r], 4, 32));
                rowmax[r] = fmaxf(rowmax[r], __shfl_xor(rowmax[r], 8, 32));
            }
            float fac[8], psum[8];
            #pragma unroll
            for (int r = 0; r < 8; ++r) {
                float mn = fmaxf(mstate[r], rowmax[r]);
                fac[r] = __expf(mstate[r] - mn);
                mstate[r] = mn;
                psum[r] = 0.0f;
            }
            // p = exp(s - m), write bf16 P to per-wave LDS scratch
            #pragma unroll
            for (int nt = 0; nt < 4; ++nt)
                #pragma unroll
                for (int r = 0; r < 8; ++r) {
                    float p = __expf(cs[nt][r] - mstate[r]);
                    psum[r] += p;
                    Pw[wave][r + hlf * 8][nt * 16 + ln15] = f2bf(p);
                }
            #pragma unroll
            for (int r = 0; r < 8; ++r) {
                psum[r] += __shfl_xor(psum[r], 1, 32);
                psum[r] += __shfl_xor(psum[r], 2, 32);
                psum[r] += __shfl_xor(psum[r], 4, 32);
                psum[r] += __shfl_xor(psum[r], 8, 32);
                lstate[r] = lstate[r] * fac[r] + psum[r];
            }
            // rescale output accumulators
            #pragma unroll
            for (int nt2 = 0; nt2 < 4; ++nt2)
                #pragma unroll
                for (int r = 0; r < 8; ++r)
                    co[nt2][r] *= fac[r];
            // same-wave LDS stores are in-order; wait for data before transposed read
            asm volatile("s_wait_dscnt 0x0" ::: "memory");
            const __bf16* prow = &Pw[wave][ln15][0];
            v16bf ap0 = mk16(ld8(prow + kb8),      ld8(prow + 16 + kb8));   // keys 0..31
            v16bf ap1 = mk16(ld8(prow + 32 + kb8), ld8(prow + 48 + kb8));   // keys 32..63
            #pragma unroll
            for (int ks = 0; ks < 2; ++ks) {
                v16bf ap = ks ? ap1 : ap0;
                #pragma unroll
                for (int nt2 = 0; nt2 < 4; ++nt2) {
                    v16bf bv = mk16(ld8(&Vt[nt2 * 16 + ln15][ks * 32 + kb8]), ld8(&Vt[nt2 * 16 + ln15][ks * 32 + 16 + kb8]));
                    co[nt2] = wmma_bf16(ap, bv, co[nt2]);
                }
            }
        }
    }

    // normalize + store context bf16 [B, S, D_MODEL]
    float inv[8];
    #pragma unroll
    for (int r = 0; r < 8; ++r) inv[r] = 1.0f / lstate[r];
    #pragma unroll
    for (int nt2 = 0; nt2 < 4; ++nt2)
        #pragma unroll
        for (int r = 0; r < 8; ++r) {
            int s = qbase + r + hlf * 8;
            int d = nt2 * 16 + ln15;
            ctx[((size_t)(b * SEQ + s)) * D_MODEL + h * DK + d] = f2bf(co[nt2][r] * inv[r]);
        }
}

// ---------------------------------------------------------------- launch
extern "C" void kernel_launch(void* const* d_in, const int* in_sizes, int n_in,
                              void* d_out, int out_size, void* d_ws, size_t ws_size,
                              hipStream_t stream) {
    const float* x    = (const float*)d_in[0];   // [4,2048,1024]
    const float* wqkv = (const float*)d_in[1];   // [3072,1024]
    const float* wo   = (const float*)d_in[2];   // [1024,1024]
    const int*   tp   = (const int*)d_in[3];     // [2048]

    char* ws = (char*)d_ws;
    size_t off = 0;
    __bf16* xbf    = (__bf16*)(ws + off); off += (size_t)MTOK * D_MODEL * 2;        // 16 MB
    __bf16* wqkvbf = (__bf16*)(ws + off); off += (size_t)3 * D_MODEL * D_MODEL * 2; // 6 MB
    __bf16* wobf   = (__bf16*)(ws + off); off += (size_t)D_MODEL * D_MODEL * 2;     // 2 MB
    __bf16* qb     = (__bf16*)(ws + off); off += (size_t)MTOK * D_MODEL * 2;        // 16 MB
    __bf16* kb     = (__bf16*)(ws + off); off += (size_t)MTOK * D_MODEL * 2;        // 16 MB
    __bf16* vb     = (__bf16*)(ws + off); off += (size_t)MTOK * D_MODEL * 2;        // 16 MB
    __bf16* ctx    = (__bf16*)(ws + off); off += (size_t)MTOK * D_MODEL * 2;        // 16 MB

    cvt_f32_bf16<<<2048, 256, 0, stream>>>(x,    xbf,    MTOK * D_MODEL);
    cvt_f32_bf16<<<1024, 256, 0, stream>>>(wqkv, wqkvbf, 3 * D_MODEL * D_MODEL);
    cvt_f32_bf16<<<512,  256, 0, stream>>>(wo,   wobf,   D_MODEL * D_MODEL);

    // QKV projection: M=8192, N=3072, K=1024
    gemm_bf16<<<dim3(3 * D_MODEL / 64, MTOK / 128), 256, 0, stream>>>(
        xbf, wqkvbf, nullptr, qb, kb, vb, 3 * D_MODEL, D_MODEL, 1);

    // RoPE on Q and K: B*H*S*32 threads
    rope_kernel<<<(BATCH * NHEADS * SEQ * 32) / 256, 256, 0, stream>>>(qb, tp);
    rope_kernel<<<(BATCH * NHEADS * SEQ * 32) / 256, 256, 0, stream>>>(kb, tp);

    // flash attention
    attn_kernel<<<dim3(SEQ / 128, BATCH * NHEADS), 256, 0, stream>>>(qb, kb, vb, ctx);

    // output projection: M=8192, N=1024, K=1024 -> fp32 d_out
    gemm_bf16<<<dim3(D_MODEL / 64, MTOK / 128), 256, 0, stream>>>(
        ctx, wobf, (float*)d_out, nullptr, nullptr, nullptr, D_MODEL, D_MODEL, 0);
}
